// EdgeModel_6691559047482
// MI455X (gfx1250) — matrix operand order repaired and checked by reading, not compiled, not run on hardware
//
#include <hip/hip_runtime.h>

// ---- problem constants (from reference) ----
#define FX   128
#define FE   64
#define FU   64
#define INC  384      // 2*FX + FE + FU
#define HID  128
#define OUTC 133
#define NP2  144      // OUTC padded to 9 WMMA n-tiles

// ---- LDS layout ----
#define W1S_STRIDE 392                    // 384 + 8 pad halfs (bank-conflict kill)
#define HB_STRIDE  136                    // 128 + 8 pad halfs
#define SMEM_W1    (HID * W1S_STRIDE)     // halfs: 50176
#define SMEM_H     (8 * 16 * HB_STRIDE)   // halfs: 17408
#define SMEM_BYTES ((SMEM_W1 + SMEM_H) * 2)  // 135168 B < 320 KB/WGP

typedef __attribute__((ext_vector_type(16))) __bf16 v16bf;
typedef __attribute__((ext_vector_type(8)))  __bf16 v8bf;
typedef __attribute__((ext_vector_type(8)))  float  v8f;

// -------- weight transpose + fp32->bf16 prep --------
__global__ void prep_weights(const float* __restrict__ W1, const float* __restrict__ W2,
                             unsigned short* __restrict__ w1t_u,
                             unsigned short* __restrict__ w2t_u) {
  __bf16* w1t = (__bf16*)w1t_u;
  __bf16* w2t = (__bf16*)w2t_u;
  int i = blockIdx.x * blockDim.x + threadIdx.x;
  if (i < INC * HID) {            // w1t[n][k] = W1[k][n]
    int n = i / INC, k = i % INC;
    w1t[i] = (__bf16)W1[k * HID + n];
  }
  if (i < NP2 * HID) {            // w2t[n][k] = W2[k][n], zero-pad n>=133
    int n = i / HID, k = i % HID;
    float v = (n < OUTC) ? W2[k * OUTC + n] : 0.0f;
    w2t[i] = (__bf16)v;
  }
}

// per-lane pointer into the virtual concatenated feature row
__device__ __forceinline__ const float* feat_ptr(int k, const float* s, const float* d,
                                                 const float* e, const float* u) {
  if (k < FX)          return s + k;
  if (k < 2 * FX)      return d + (k - FX);
  if (k < 2 * FX + FE) return e + (k - 2 * FX);
  return u + (k - 2 * FX - FE);
}

// build bf16 A-fragment from two 8-float global chunks (K=kb..kb+7, K=kb+16..kb+23)
__device__ __forceinline__ v16bf a_frag_f32(const float* c0, const float* c1) {
  float4 x0 = *(const float4*)(c0);
  float4 x1 = *(const float4*)(c0 + 4);
  float4 y0 = *(const float4*)(c1);
  float4 y1 = *(const float4*)(c1 + 4);
  v16bf a;
  a[0]  = (__bf16)x0.x; a[1]  = (__bf16)x0.y; a[2]  = (__bf16)x0.z; a[3]  = (__bf16)x0.w;
  a[4]  = (__bf16)x1.x; a[5]  = (__bf16)x1.y; a[6]  = (__bf16)x1.z; a[7]  = (__bf16)x1.w;
  a[8]  = (__bf16)y0.x; a[9]  = (__bf16)y0.y; a[10] = (__bf16)y0.z; a[11] = (__bf16)y0.w;
  a[12] = (__bf16)y1.x; a[13] = (__bf16)y1.y; a[14] = (__bf16)y1.z; a[15] = (__bf16)y1.w;
  return a;
}

// 16-half fragment from two 8-half chunks at p and p+secondOff (16B aligned)
__device__ __forceinline__ v16bf frag16(const __bf16* p, int secondOff) {
  v8bf lo = *(const v8bf*)(p);
  v8bf hi = *(const v8bf*)(p + secondOff);
  return __builtin_shufflevector(lo, hi, 0, 1, 2, 3, 4, 5, 6, 7,
                                         8, 9, 10, 11, 12, 13, 14, 15);
}

// -------- fused edge-MLP: concat -> GEMM1 -> ReLU -> GEMM2 --------
__global__ __launch_bounds__(256, 2)
void edge_mlp(const float* __restrict__ src, const float* __restrict__ dst,
              const float* __restrict__ ea,  const float* __restrict__ u,
              const long long* __restrict__ batch,
              const float* __restrict__ b1,  const float* __restrict__ b2,
              const unsigned short* __restrict__ w1t_u,
              const unsigned short* __restrict__ w2t_u,
              float* __restrict__ out, int E) {
  extern __shared__ char smem[];
  __bf16* w1s  = (__bf16*)smem;              // [128][392] W1^T bf16
  __bf16* hbuf = (__bf16*)smem + SMEM_W1;    // [8 waves][16][136] hidden bf16

  const __bf16* w1t = (const __bf16*)w1t_u;
  const __bf16* w2t = (const __bf16*)w2t_u;

  const int tid = threadIdx.x;

  // stage W1^T into LDS (8B chunks, coalesced global reads)
  for (int i = tid * 4; i < INC * HID; i += 256 * 4) {
    int n = i / INC, k = i % INC;
    *(uint2*)(w1s + n * W1S_STRIDE + k) = *(const uint2*)(w1t + i);
  }
  __syncthreads();

  const int lane = tid & 31;
  const int wave = tid >> 5;
  const int half = lane >> 4;   // 0: lanes 0-15, 1: lanes 16-31
  const int l16  = lane & 15;

  const long eBase = (long)blockIdx.x * 128 + wave * 16;
  long eRow = eBase + l16;                 // A-fragment row owned by this lane
  if (eRow >= E) eRow = (long)E - 1;       // clamp; OOB rows masked at store

  const float* sRow = src + eRow * FX;
  const float* dRow = dst + eRow * FX;
  const float* aRow = ea  + eRow * FE;
  const float* uRow = u + (long)batch[eRow] * FU;

  const v8f vzero = {0.f, 0.f, 0.f, 0.f, 0.f, 0.f, 0.f, 0.f};

  // ===== GEMM1: [16x384] @ [384x128], 12 k-tiles x 8 n-tiles =====
  v8f acc[8];
  #pragma unroll
  for (int j = 0; j < 8; ++j) acc[j] = vzero;

  #pragma unroll
  for (int t = 0; t < 12; ++t) {
    const int kb = 32 * t + half * 8;      // A chunks: [kb,kb+8) and [kb+16,kb+24)
    const float* c0 = feat_ptr(kb,      sRow, dRow, aRow, uRow);
    const float* c1 = feat_ptr(kb + 16, sRow, dRow, aRow, uRow);
    v16bf afrag = a_frag_f32(c0, c1);
    const int k0 = 32 * t + half * 8;      // B chunks: [k0,k0+8) and [k0+16,k0+24)
    const __bf16* bbase = w1s + l16 * W1S_STRIDE + k0;
    // software pipeline: fetch B[j+1] before WMMA on B[j] so the ds_load
    // overlaps the matrix op instead of a full s_wait_dscnt 0 stall
    v16bf bcur = frag16(bbase, 16);
    #pragma unroll
    for (int j = 0; j < 8; ++j) {
      v16bf bnxt = bcur;
      if (j < 7) bnxt = frag16(bbase + (j + 1) * 16 * W1S_STRIDE, 16);
      acc[j] = __builtin_amdgcn_wmma_f32_16x16x32_bf16(
          false, afrag, false, bcur, (short)0, acc[j], false, false);
      bcur = bnxt;
    }
  }

  // ===== epilogue 1: bias + ReLU, transpose C-layout -> A-layout via LDS =====
  __bf16* hb = hbuf + wave * 16 * HB_STRIDE;
  const int mb = half * 8;                 // C rows: 0-7 (lo lanes) / 8-15 (hi lanes)
  #pragma unroll
  for (int j = 0; j < 8; ++j) {
    const int col = j * 16 + l16;
    const float bias = b1[col];
    #pragma unroll
    for (int r = 0; r < 8; ++r) {
      float v = acc[j][r] + bias;
      hb[(mb + r) * HB_STRIDE + col] = (__bf16)(v > 0.f ? v : 0.f);
    }
  }
  // per-wave LDS buffer: same-wave DS ops are in program order, no barrier needed

  // ===== GEMM2: [16x128] @ [128x144], 4 k-tiles x 9 n-tiles =====
  v8f acc2[9];
  #pragma unroll
  for (int j = 0; j < 9; ++j) acc2[j] = vzero;

  #pragma unroll
  for (int t = 0; t < 4; ++t) {
    const int kb = 32 * t + half * 8;
    const __bf16* ap = hb + l16 * HB_STRIDE + kb;
    v16bf afrag = frag16(ap, 16);          // chunks at kb and kb+16
    const int k0 = 32 * t + half * 8;
    const __bf16* b2base = w2t + l16 * HID + k0;   // L2-resident
    v16bf bcur = frag16(b2base, 16);
    #pragma unroll
    for (int j = 0; j < 9; ++j) {
      v16bf bnxt = bcur;
      if (j < 8) bnxt = frag16(b2base + (j + 1) * 16 * HID, 16);
      acc2[j] = __builtin_amdgcn_wmma_f32_16x16x32_bf16(
          false, afrag, false, bcur, (short)0, acc2[j], false, false);
      bcur = bnxt;
    }
  }

  // ===== epilogue 2: bias + masked store of 133 valid columns =====
  #pragma unroll
  for (int j = 0; j < 9; ++j) {
    const int col = j * 16 + l16;
    if (col < OUTC) {
      const float bias = b2[col];
      #pragma unroll
      for (int r = 0; r < 8; ++r) {
        const long e = eBase + mb + r;
        if (e < E) out[e * OUTC + col] = acc2[j][r] + bias;
      }
    }
  }
}

extern "C" void kernel_launch(void* const* d_in, const int* in_sizes, int n_in,
                              void* d_out, int out_size, void* d_ws, size_t ws_size,
                              hipStream_t stream) {
  (void)n_in; (void)out_size; (void)ws_size;
  const float*     src   = (const float*)d_in[0];
  const float*     dst   = (const float*)d_in[1];
  const float*     ea    = (const float*)d_in[2];
  const float*     u     = (const float*)d_in[3];
  const long long* batch = (const long long*)d_in[4];   // int64
  const float*     W1    = (const float*)d_in[5];
  const float*     b1    = (const float*)d_in[6];
  const float*     W2    = (const float*)d_in[7];
  const float*     b2    = (const float*)d_in[8];
  float*           out   = (float*)d_out;

  const int E = in_sizes[0] / FX;

  unsigned short* w1t = (unsigned short*)d_ws;          // 384*128 bf16
  unsigned short* w2t = w1t + INC * HID;                // 144*128 bf16

  // allow 132 KB dynamic LDS (deterministic; not a stream op, capture-safe)
  hipFuncSetAttribute((const void*)edge_mlp,
                      hipFuncAttributeMaxDynamicSharedMemorySize, SMEM_BYTES);

  prep_weights<<<(INC * HID + 255) / 256, 256, 0, stream>>>(W1, W2, w1t, w2t);

  const int nBlocks = (E + 127) / 128;   // 128 edges / block (8 waves x 16)
  edge_mlp<<<nBlocks, 256, SMEM_BYTES, stream>>>(src, dst, ea, u, batch,
                                                 b1, b2, w1t, w2t, out, E);
}
